// Attention_70033736728830
// MI455X (gfx1250) — matrix operand order repaired and checked by reading, not compiled, not run on hardware
//
#include <hip/hip_runtime.h>
#include <hip/hip_bf16.h>
#include <math.h>

// Problem constants (from setup_inputs)
#define B_SZ 128
#define NTOK 388      // Nt + Ns
#define CCH  768      // C
#define C3   2304     // 3C
#define NH   12       // heads
#define HD   64       // head dim
#define NT   128      // mosaic tokens
#define NS_  260      // s tokens
#define NPAD 416      // keys padded to multiple of 32 (388 -> 416)
#define SCALE_ 0.125f // 1/sqrt(64)

typedef __attribute__((ext_vector_type(16))) _Float16 v16h;
typedef __attribute__((ext_vector_type(8)))  float    v8f;
typedef __attribute__((ext_vector_type(4)))  unsigned int v4u;
typedef __attribute__((ext_vector_type(4)))  int v4i;

// ---------------- gfx1250 async global->LDS (guarded) -----------------------
#if defined(__has_builtin)
#if __has_builtin(__builtin_amdgcn_global_load_async_to_lds_b128)
#define HAVE_ASYNC_LDS 1
#endif
#if __has_builtin(__builtin_amdgcn_s_wait_asynccnt)
#define HAVE_WAIT_ASYNC 1
#endif
#endif

typedef __attribute__((address_space(1))) v4i as1_v4i;
typedef __attribute__((address_space(3))) v4i as3_v4i;

static __device__ __forceinline__ void async_copy16(const _Float16* g,
                                                    _Float16* l) {
#ifdef HAVE_ASYNC_LDS
  __builtin_amdgcn_global_load_async_to_lds_b128((as1_v4i*)g, (as3_v4i*)l, 0, 0);
#else
  *(v4u*)l = *(const v4u*)g;  // sync fallback; barriers provide ordering
#endif
}
static __device__ __forceinline__ void wait_async0() {
#ifdef HAVE_ASYNC_LDS
#ifdef HAVE_WAIT_ASYNC
  __builtin_amdgcn_s_wait_asynccnt(0);
#else
  asm volatile("s_wait_asynccnt 0x0" ::: "memory");
#endif
#endif
}
static __device__ __forceinline__ void wait_async1() {
#ifdef HAVE_ASYNC_LDS
#ifdef HAVE_WAIT_ASYNC
  __builtin_amdgcn_s_wait_asynccnt(1);
#else
  asm volatile("s_wait_asynccnt 0x1" ::: "memory");
#endif
#endif
}

// ---------------- WMMA fragment helpers (CDNA5 16x16x32 f16 layouts) --------
// A (16x32, M x K): lane m (0-15) row m, K = {hi*8..+7} U {16+hi*8..+7}
static __device__ __forceinline__ v16h frag_a(const _Float16* tile, int lda) {
  int lane = threadIdx.x & 31;
  const _Float16* p = tile + (long)(lane & 15) * lda + (lane >> 4) * 8;
  v16h r;
  ((v4u*)&r)[0] = *(const v4u*)(p);
  ((v4u*)&r)[1] = *(const v4u*)(p + 16);
  return r;
}
// B (32x16, K x N) fed from a row-major [N,K] matrix ("A @ B^T" form):
// lane n holds column n, K = {hi*8..+7} U {16+hi*8..+7}
static __device__ __forceinline__ v16h frag_b(const _Float16* tile, int ldb) {
  int lane = threadIdx.x & 31;
  const _Float16* p = tile + (long)(lane & 15) * ldb + (lane >> 4) * 8;
  v16h r;
  ((v4u*)&r)[0] = *(const v4u*)(p);
  ((v4u*)&r)[1] = *(const v4u*)(p + 16);
  return r;
}
static __device__ __forceinline__ v8f wmma_f16(v16h a, v16h b, v8f c) {
  return __builtin_amdgcn_wmma_f32_16x16x32_f16(false, a, false, b, (short)0, c,
                                                false, false);
}

// ---------------- elementwise converts / transposes -------------------------
__global__ void cvt_kern(const float* __restrict__ s, _Float16* __restrict__ d,
                         long n) {
  long i = (long)blockIdx.x * 256 + threadIdx.x;
  if (i < n) d[i] = (_Float16)s[i];
}

// vT[b,h][c][key] = v section of qkv ; zero-pad keys >= NTOK
__global__ void build_vT_kern(const _Float16* __restrict__ qkv,
                              _Float16* __restrict__ vT) {
  long i = (long)blockIdx.x * 256 + threadIdx.x;
  long total = (long)B_SZ * NH * HD * NPAD;
  if (i >= total) return;
  int key = (int)(i % NPAD);
  long r = i / NPAD;
  int c = (int)(r % HD);
  long bh = r / HD;
  int h = (int)(bh % NH);
  int b = (int)(bh / NH);
  _Float16 v = (_Float16)0.f;
  if (key < NTOK)
    v = qkv[((long)b * NTOK + key) * C3 + 2 * CCH + (long)h * HD + c];
  vT[i] = v;
}

// memT[b,h][e][dd] = (f16) mem[b,h][dd][e]
__global__ void build_memT_kern(const float* __restrict__ mem,
                                _Float16* __restrict__ mT) {
  long i = (long)blockIdx.x * 256 + threadIdx.x;
  long total = (long)B_SZ * NH * HD * HD;
  if (i >= total) return;
  int dd = (int)(i & 63);
  long r = i >> 6;
  int e = (int)(r & 63);
  long bh = r >> 6;
  mT[i] = (_Float16)mem[bh * HD * HD + (long)dd * HD + e];
}

// ---------------- cooperative GEMM: D = A[M,K] @ W[Nout,K]^T + bias ---------
// 256 threads = 8 waves; block computes 128(M) x 64(N); wave w owns rows
// [128*tile + 16w, +16). The 64x32 weight slab is staged in LDS (8x reuse)
// via double-buffered async global->LDS copies; A fragments are register
// double-buffered so global loads overlap the WMMA pipe.
template <typename OutT>
__global__ __launch_bounds__(256) void gemm_bias_kern(
    const _Float16* __restrict__ A, int lda, long bsA, int rowsPerBatch,
    int mtilesPerBatch, const _Float16* __restrict__ W, int K,
    const float* __restrict__ bias, OutT* __restrict__ D, int ldd, long bsD) {
  int bidx = blockIdx.x;
  int batch = bidx / mtilesPerBatch;
  int tin = bidx % mtilesPerBatch;  // 128-row tile index within batch
  int n0 = blockIdx.y * 64;
  int tid = threadIdx.x;
  int wave = tid >> 5;

  __shared__ __align__(16) _Float16 Wb[2][64 * 32];
  // staging map: thread t -> weight row n = t>>2, 8-half chunk (t&3)*8
  int srow = tid >> 2, schunk = (tid & 3) * 8;
  const _Float16* Wg = W + (long)(n0 + srow) * K + schunk;
  _Float16* Wl = &Wb[0][0] + srow * 32 + schunk;

  const _Float16* Ab =
      A + (long)batch * bsA + ((long)tin * 128 + (long)wave * 16) * lda;

  int nk = K / 32;
  // prologue: stage k-slabs 0 and 1
  async_copy16(Wg + 0, Wl);
  async_copy16(Wg + 32, Wl + 64 * 32);
  v16h af = frag_a(Ab, lda);
  v8f a0 = {}, a1 = {}, a2 = {}, a3 = {};
  for (int kt = 0; kt < nk; ++kt) {
    if (kt + 1 < nk) wait_async1(); else wait_async0();
    __syncthreads();  // slab kt visible to all 8 waves
    v16h afn = af;
    if (kt + 1 < nk) afn = frag_a(Ab + (kt + 1) * 32, lda);
    const _Float16* bb = &Wb[kt & 1][0];
    a0 = wmma_f16(af, frag_b(bb + 0 * 32, 32), a0);
    a1 = wmma_f16(af, frag_b(bb + 16 * 32, 32), a1);
    a2 = wmma_f16(af, frag_b(bb + 32 * 32, 32), a2);
    a3 = wmma_f16(af, frag_b(bb + 48 * 32, 32), a3);
    __syncthreads();  // all waves done reading slab kt
    if (kt + 2 < nk) async_copy16(Wg + (kt + 2) * 32, Wl + (kt & 1) * (64 * 32));
    af = afn;
  }
  int lane = tid & 31, hi = lane >> 4, nl = lane & 15;
  float b0 = bias[n0 + nl], b1 = bias[n0 + 16 + nl];
  float b2 = bias[n0 + 32 + nl], b3 = bias[n0 + 48 + nl];
#pragma unroll
  for (int g = 0; g < 8; ++g) {
    int row = tin * 128 + wave * 16 + g + 8 * hi;
    if (row < rowsPerBatch) {
      OutT* dr = D + (long)batch * bsD + (long)row * ldd + n0 + nl;
      dr[0]  = (OutT)(a0[g] + b0);
      dr[16] = (OutT)(a1[g] + b1);
      dr[32] = (OutT)(a2[g] + b2);
      dr[48] = (OutT)(a3[g] + b3);
    }
  }
}

// ---------------- fused attention, mosaic branch (Nt x Nt) ------------------
__global__ __launch_bounds__(32) void attn_mt_kern(
    const _Float16* __restrict__ qkv, const _Float16* __restrict__ vT,
    _Float16* __restrict__ y) {
  int id = blockIdx.x;
  int qt = id & 7;  // 8 q-tiles of 16
  int bh = id >> 3;
  int h = bh % NH, b = bh / NH;
  __shared__ __align__(16) float    sc[16][132];
  __shared__ __align__(16) _Float16 pr[16][128];
  int lane = threadIdx.x & 31, hi = lane >> 4, nl = lane & 15;
  const _Float16* qb = qkv + ((long)b * NTOK + qt * 16) * C3 + (long)h * HD;
  const _Float16* kb = qkv + (long)b * NTOK * C3 + CCH + (long)h * HD;
  // scores (register double-buffered B fragments)
  v16h qa0 = frag_a(qb + 0, C3), qa1 = frag_a(qb + 32, C3);
  v16h kb0 = frag_b(kb + 0, C3), kb1 = frag_b(kb + 32, C3);
  for (int kt = 0; kt < 8; ++kt) {
    v16h nb0 = kb0, nb1 = kb1;
    if (kt + 1 < 8) {
      nb0 = frag_b(kb + (long)(kt + 1) * 16 * C3 + 0, C3);
      nb1 = frag_b(kb + (long)(kt + 1) * 16 * C3 + 32, C3);
    }
    v8f s = {};
    s = wmma_f16(qa0, kb0, s);
    s = wmma_f16(qa1, kb1, s);
#pragma unroll
    for (int g = 0; g < 8; ++g) sc[g + 8 * hi][kt * 16 + nl] = s[g] * SCALE_;
    kb0 = nb0; kb1 = nb1;
  }
  __syncthreads();
  if (lane < 16) {
    float m = -1e30f;
    for (int j = 0; j < 128; ++j) m = fmaxf(m, sc[lane][j]);
    float sum = 0.f;
    for (int j = 0; j < 128; ++j) {
      float e = expf(sc[lane][j] - m);
      sc[lane][j] = e;
      sum += e;
    }
    float inv = 1.f / sum;
    for (int j = 0; j < 128; ++j) pr[lane][j] = (_Float16)(sc[lane][j] * inv);
  }
  __syncthreads();
  const _Float16* vb = vT + (long)bh * HD * NPAD;
  v8f o0 = {}, o1 = {}, o2 = {}, o3 = {};
  v16h pa = frag_a(&pr[0][0], 128);
  v16h w0 = frag_b(vb + 0 * NPAD, NPAD), w1 = frag_b(vb + 16 * NPAD, NPAD);
  v16h w2 = frag_b(vb + 32 * NPAD, NPAD), w3 = frag_b(vb + 48 * NPAD, NPAD);
  for (int kt = 0; kt < 4; ++kt) {
    int k0 = kt * 32;
    v16h pan = pa, n0f = w0, n1f = w1, n2f = w2, n3f = w3;
    if (kt + 1 < 4) {
      int kn = k0 + 32;
      pan = frag_a(&pr[0][0] + kn, 128);
      n0f = frag_b(vb + 0 * NPAD + kn, NPAD);
      n1f = frag_b(vb + 16 * NPAD + kn, NPAD);
      n2f = frag_b(vb + 32 * NPAD + kn, NPAD);
      n3f = frag_b(vb + 48 * NPAD + kn, NPAD);
    }
    o0 = wmma_f16(pa, w0, o0);
    o1 = wmma_f16(pa, w1, o1);
    o2 = wmma_f16(pa, w2, o2);
    o3 = wmma_f16(pa, w3, o3);
    pa = pan; w0 = n0f; w1 = n1f; w2 = n2f; w3 = n3f;
  }
#pragma unroll
  for (int g = 0; g < 8; ++g) {
    int tok = qt * 16 + g + 8 * hi;
    _Float16* yr = y + ((long)b * NTOK + tok) * CCH + (long)h * HD + nl;
    yr[0]  = (_Float16)o0[g];
    yr[16] = (_Float16)o1[g];
    yr[32] = (_Float16)o2[g];
    yr[48] = (_Float16)o3[g];
  }
}

// ------- fused attention, s branch + att_mem read + beta blend --------------
__global__ __launch_bounds__(32) void attn_s_kern(
    const _Float16* __restrict__ qkv, const _Float16* __restrict__ vT,
    const _Float16* __restrict__ memT, const float* __restrict__ z,
    const float* __restrict__ betas, _Float16* __restrict__ y) {
  int id = blockIdx.x;
  int qt = id % 17;  // 17 q-tiles cover Ns=260
  int bh = id / 17;
  int h = bh % NH, b = bh / NH;
  __shared__ __align__(16) float    sc[16][404];
  __shared__ __align__(16) _Float16 pr[16][NPAD];
  __shared__ __align__(16) _Float16 sq[16][HD];
  __shared__ float den[16];
  int lane = threadIdx.x & 31, hi = lane >> 4, nl = lane & 15;
  const _Float16* qb =
      qkv + ((long)b * NTOK + NT + qt * 16) * C3 + (long)h * HD;
  const _Float16* kb = qkv + (long)b * NTOK * C3 + CCH + (long)h * HD;
  v16h qa0 = frag_a(qb + 0, C3), qa1 = frag_a(qb + 32, C3);
  v16h kb0 = frag_b(kb + 0, C3), kb1 = frag_b(kb + 32, C3);
  for (int kt = 0; kt < 25; ++kt) {  // 25*16 = 400 >= 388 keys
    v16h nb0 = kb0, nb1 = kb1;
    if (kt + 1 < 25) {
      nb0 = frag_b(kb + (long)(kt + 1) * 16 * C3 + 0, C3);
      nb1 = frag_b(kb + (long)(kt + 1) * 16 * C3 + 32, C3);
    }
    v8f s = {};
    s = wmma_f16(qa0, kb0, s);
    s = wmma_f16(qa1, kb1, s);
#pragma unroll
    for (int g = 0; g < 8; ++g) sc[g + 8 * hi][kt * 16 + nl] = s[g] * SCALE_;
    kb0 = nb0; kb1 = nb1;
  }
  // sigma_q = elu(q)+1  (x>0 ? x+1 : exp(x)) into LDS as f16
  for (int idx = lane; idx < 16 * HD; idx += 32) {
    int r = idx >> 6, dd = idx & 63;
    float qv = (float)qb[(long)r * C3 + dd];
    float sig = qv > 0.f ? qv + 1.f : expf(qv);
    sq[r][dd] = (_Float16)sig;
  }
  __syncthreads();
  if (lane < 16) {
    float m = -1e30f;
    for (int j = 0; j < NTOK; ++j) m = fmaxf(m, sc[lane][j]);
    float sum = 0.f;
    for (int j = 0; j < NTOK; ++j) {
      float e = expf(sc[lane][j] - m);
      sc[lane][j] = e;
      sum += e;
    }
    float inv = 1.f / sum;
    for (int j = 0; j < NTOK; ++j) pr[lane][j] = (_Float16)(sc[lane][j] * inv);
    for (int j = NTOK; j < NPAD; ++j) pr[lane][j] = (_Float16)0.f;
    const float* zp = z + (long)bh * HD;
    float ds = 0.f;
    for (int dd = 0; dd < HD; ++dd) ds += (float)sq[lane][dd] * zp[dd];
    den[lane] = ds;
  }
  __syncthreads();
  // x_s = softmax(qk) @ v  (K = 416, register double-buffered)
  const _Float16* vb = vT + (long)bh * HD * NPAD;
  v8f o0 = {}, o1 = {}, o2 = {}, o3 = {};
  {
    v16h pa = frag_a(&pr[0][0], NPAD);
    v16h w0 = frag_b(vb + 0 * NPAD, NPAD), w1 = frag_b(vb + 16 * NPAD, NPAD);
    v16h w2 = frag_b(vb + 32 * NPAD, NPAD), w3 = frag_b(vb + 48 * NPAD, NPAD);
    const int KT = NPAD / 32;  // 13
    for (int kt = 0; kt < KT; ++kt) {
      v16h pan = pa, n0f = w0, n1f = w1, n2f = w2, n3f = w3;
      if (kt + 1 < KT) {
        int kn = (kt + 1) * 32;
        pan = frag_a(&pr[0][0] + kn, NPAD);
        n0f = frag_b(vb + 0 * NPAD + kn, NPAD);
        n1f = frag_b(vb + 16 * NPAD + kn, NPAD);
        n2f = frag_b(vb + 32 * NPAD + kn, NPAD);
        n3f = frag_b(vb + 48 * NPAD + kn, NPAD);
      }
      o0 = wmma_f16(pa, w0, o0);
      o1 = wmma_f16(pa, w1, o1);
      o2 = wmma_f16(pa, w2, o2);
      o3 = wmma_f16(pa, w3, o3);
      pa = pan; w0 = n0f; w1 = n1f; w2 = n2f; w3 = n3f;
    }
  }
  // att_mem numerator = sigma_q @ mem  (memT rows = columns of mem)
  const _Float16* mb = memT + (long)bh * HD * HD;
  v8f m0 = {}, m1 = {}, m2 = {}, m3 = {};
  for (int k0 = 0; k0 < HD; k0 += 32) {
    v16h af = frag_a(&sq[0][0] + k0, HD);
    m0 = wmma_f16(af, frag_b(mb + 0 * HD + k0, HD), m0);
    m1 = wmma_f16(af, frag_b(mb + 16 * HD + k0, HD), m1);
    m2 = wmma_f16(af, frag_b(mb + 32 * HD + k0, HD), m2);
    m3 = wmma_f16(af, frag_b(mb + 48 * HD + k0, HD), m3);
  }
  const float* bp = betas + (long)h * HD;
  float be0 = 1.f / (1.f + expf(-bp[nl]));
  float be1 = 1.f / (1.f + expf(-bp[16 + nl]));
  float be2 = 1.f / (1.f + expf(-bp[32 + nl]));
  float be3 = 1.f / (1.f + expf(-bp[48 + nl]));
#pragma unroll
  for (int g = 0; g < 8; ++g) {
    int row = g + 8 * hi;
    int qrow = qt * 16 + row;
    if (qrow < NS_) {
      float dn = den[row];
      _Float16* yr =
          y + ((long)b * NTOK + NT + qrow) * CCH + (long)h * HD + nl;
      yr[0]  = (_Float16)(be0 * (m0[g] / dn) + (1.f - be0) * o0[g]);
      yr[16] = (_Float16)(be1 * (m1[g] / dn) + (1.f - be1) * o1[g]);
      yr[32] = (_Float16)(be2 * (m2[g] / dn) + (1.f - be2) * o2[g]);
      yr[48] = (_Float16)(be3 * (m3[g] / dn) + (1.f - be3) * o3[g]);
    }
  }
}

// ------- memory state update: mem_new, z_new (small, f32 VALU) --------------
__global__ __launch_bounds__(256) void mem_update_kern(
    const _Float16* __restrict__ ksm, const _Float16* __restrict__ vsm,
    const float* __restrict__ mem, const float* __restrict__ z,
    float* __restrict__ mem_out, float* __restrict__ z_out) {
  int bh = blockIdx.x;
  int h = bh % NH, b = bh / NH;
  __shared__ float sMem[HD * HD];
  __shared__ float sZ[HD];
  __shared__ float sSig[HD];
  int t = threadIdx.x;  // 256 threads
  const float* mp = mem + (long)bh * HD * HD;
  for (int i = t; i < HD * HD; i += 256) sMem[i] = mp[i];
  if (t < HD) sZ[t] = z[(long)bh * HD + t];
  __syncthreads();
  float acc[16];
#pragma unroll
  for (int j = 0; j < 16; ++j) acc[j] = 0.f;
  float zsum = 0.f;
  int e = t & 63, dg = t >> 6;  // thread owns column e, rows dg+4*j
  for (int n = 0; n < NS_; ++n) {
    const _Float16* kr = ksm + ((long)b * NS_ + n) * CCH + (long)h * HD;
    if (t < HD) {
      float x = (float)kr[t];
      float s = x > 0.f ? x + 1.f : expf(x);
      sSig[t] = s;
      zsum += s;
    }
    __syncthreads();
    float num = 0.f, dnv = 0.f;
    for (int dd = 0; dd < HD; ++dd) {
      float s = sSig[dd];
      num += s * sMem[dd * HD + e];
      dnv += s * sZ[dd];
    }
    float vv = (float)vsm[((long)b * NS_ + n) * CCH + (long)h * HD + e];
    float diff = vv - num / dnv;
#pragma unroll
    for (int j = 0; j < 16; ++j) acc[j] += sSig[dg + 4 * j] * diff;
    __syncthreads();
  }
#pragma unroll
  for (int j = 0; j < 16; ++j) {
    int dd = dg + 4 * j;
    mem_out[(long)bh * HD * HD + (long)dd * HD + e] = sMem[dd * HD + e] + acc[j];
  }
  if (t < HD) z_out[(long)bh * HD + t] = sZ[t] + zsum;
}

// ---------------------------------------------------------------------------
extern "C" void kernel_launch(void* const* d_in, const int* in_sizes, int n_in,
                              void* d_out, int out_size, void* d_ws,
                              size_t ws_size, hipStream_t stream) {
  const float* x      = (const float*)d_in[0];
  const float* mem    = (const float*)d_in[1];
  const float* z      = (const float*)d_in[2];
  const float* qkv_w  = (const float*)d_in[3];
  const float* qkv_b  = (const float*)d_in[4];
  const float* proj_w = (const float*)d_in[5];
  const float* proj_b = (const float*)d_in[6];
  const float* memk_w = (const float*)d_in[7];
  const float* memk_b = (const float*)d_in[8];
  const float* memv_w = (const float*)d_in[9];
  const float* memv_b = (const float*)d_in[10];
  const float* betas  = (const float*)d_in[11];

  float* out     = (float*)d_out;
  float* mem_new = out + (long)B_SZ * NTOK * CCH;
  float* z_new   = mem_new + (long)B_SZ * NH * HD * HD;

  char* p = (char*)d_ws;
  auto alloc = [&](size_t bytes) -> char* {
    char* r = p;
    p += (bytes + 255) & ~(size_t)255;
    return r;
  };
  _Float16* x_h   = (_Float16*)alloc((size_t)B_SZ * NTOK * CCH * 2);
  _Float16* qw_h  = (_Float16*)alloc((size_t)C3 * CCH * 2);
  _Float16* kw_h  = (_Float16*)alloc((size_t)CCH * CCH * 2);
  _Float16* vw_h  = (_Float16*)alloc((size_t)CCH * CCH * 2);
  _Float16* pw_h  = (_Float16*)alloc((size_t)CCH * CCH * 2);
  _Float16* qkv_h = (_Float16*)alloc(((size_t)B_SZ * NTOK + 128) * C3 * 2); // +128 row slack
  _Float16* vT    = (_Float16*)alloc((size_t)B_SZ * NH * HD * NPAD * 2);
  _Float16* mT    = (_Float16*)alloc((size_t)B_SZ * NH * HD * HD * 2);
  _Float16* ks_h  = (_Float16*)alloc((size_t)B_SZ * NS_ * CCH * 2);
  _Float16* vs_h  = (_Float16*)alloc((size_t)B_SZ * NS_ * CCH * 2);
  _Float16* y_h   = (_Float16*)alloc((size_t)B_SZ * NTOK * CCH * 2);

  auto cvtL = [&](const float* s, _Float16* d, long n) {
    cvt_kern<<<dim3((unsigned)((n + 255) / 256)), dim3(256), 0, stream>>>(s, d, n);
  };
  cvtL(x, x_h, (long)B_SZ * NTOK * CCH);
  cvtL(qkv_w, qw_h, (long)C3 * CCH);
  cvtL(memk_w, kw_h, (long)CCH * CCH);
  cvtL(memv_w, vw_h, (long)CCH * CCH);
  cvtL(proj_w, pw_h, (long)CCH * CCH);

  const int MT128 = (B_SZ * NTOK) / 128;  // 388 (exact)

  // qkv = x @ qkv_w^T + qkv_b  (f16 out)
  gemm_bias_kern<_Float16><<<dim3(MT128, C3 / 64), dim3(256), 0, stream>>>(
      x_h, CCH, 0L, B_SZ * NTOK, MT128, qw_h, CCH, qkv_b, qkv_h, C3, 0L);

  {
    long n = (long)B_SZ * NH * HD * NPAD;
    build_vT_kern<<<dim3((unsigned)((n + 255) / 256)), dim3(256), 0, stream>>>(qkv_h, vT);
  }
  {
    long n = (long)B_SZ * NH * HD * HD;
    build_memT_kern<<<dim3((unsigned)((n + 255) / 256)), dim3(256), 0, stream>>>(mem, mT);
  }

  // k_s_mem / v_s_mem: rows are k/v sections of tokens Nt..N per batch
  // ceil(260/128) = 3 row-tiles per batch (stores guarded)
  gemm_bias_kern<_Float16><<<dim3(B_SZ * 3, CCH / 64), dim3(256), 0, stream>>>(
      qkv_h + (long)NT * C3 + CCH, C3, (long)NTOK * C3, NS_, 3, kw_h, CCH,
      memk_b, ks_h, CCH, (long)NS_ * CCH);
  gemm_bias_kern<_Float16><<<dim3(B_SZ * 3, CCH / 64), dim3(256), 0, stream>>>(
      qkv_h + (long)NT * C3 + 2 * CCH, C3, (long)NTOK * C3, NS_, 3, vw_h, CCH,
      memv_b, vs_h, CCH, (long)NS_ * CCH);

  // fused attention branches
  attn_mt_kern<<<dim3(B_SZ * NH * 8), dim3(32), 0, stream>>>(qkv_h, vT, y_h);
  attn_s_kern<<<dim3(B_SZ * NH * 17), dim3(32), 0, stream>>>(qkv_h, vT, mT, z,
                                                             betas, y_h);

  // memory-state update (mem_new, z_new)
  mem_update_kern<<<dim3(B_SZ * NH), dim3(256), 0, stream>>>(ks_h, vs_h, mem, z,
                                                             mem_new, z_new);

  // out = y @ proj_w^T + proj_b  (f32 out, straight to d_out)
  gemm_bias_kern<float><<<dim3(MT128, CCH / 64), dim3(256), 0, stream>>>(
      y_h, CCH, 0L, B_SZ * NTOK, MT128, pw_h, CCH, proj_b, out, CCH, 0L);

  (void)in_sizes; (void)n_in; (void)out_size; (void)ws_size;
}
